// RopeMultiHeadAttentionGQA_80272938762893
// MI455X (gfx1250) — matrix operand order repaired and checked by reading, not compiled, not run on hardware
//
#include <hip/hip_runtime.h>

// ---------------------------------------------------------------------------
// RoPE + GQA skip-window attention for MI455X (gfx1250), wave32 + WMMA + TDM.
// GEMMs (QKV proj, output proj) use v_wmma_f32_16x16x32_f16 with the B-tile
// staged into LDS by the Tensor Data Mover (tensor_load_to_lds), double
// buffered against compute. Attention is a masked flash kernel exploiting the
// skip-window sparsity. Softmax/RoPE run in fp32 VALU.
// ---------------------------------------------------------------------------

typedef __attribute__((ext_vector_type(16))) _Float16 v16h;
typedef __attribute__((ext_vector_type(8)))  _Float16 v8h;
typedef __attribute__((ext_vector_type(8)))  float    v8f;
typedef __attribute__((ext_vector_type(4)))  unsigned int v4u;
typedef __attribute__((ext_vector_type(8)))  int      v8i;
typedef __attribute__((ext_vector_type(4)))  int      v4i;

#ifndef __has_builtin
#define __has_builtin(x) 0
#endif
#if __has_builtin(__builtin_amdgcn_tensor_load_to_lds) && \
    __has_builtin(__builtin_amdgcn_s_wait_tensorcnt)
#define HAVE_TDM 1
#else
#define HAVE_TDM 0
#endif

static constexpr int Bn   = 2;
static constexpr int Tn   = 2048;
static constexpr int Dn   = 1024;
static constexpr int Hn   = 16;
static constexpr int HKVn = 4;
static constexpr int HDc  = 64;
static constexpr int NREP = Hn / HKVn;              // 4
static constexpr int HALF_WIN = 12;                 // SKIP_WINDOW/2
static constexpr int GLOB = 8;
static constexpr float NEG_INF = -1.0e9f;
static constexpr float SCALE   = 0.125f;            // 1/sqrt(64)

__device__ __forceinline__ v16h cat8(v8h lo, v8h hi) {
  return __builtin_shufflevector(lo, hi, 0,1,2,3,4,5,6,7,8,9,10,11,12,13,14,15);
}

__device__ __forceinline__ v8f wmma_f16(v16h a, v16h b, v8f c) {
  // D = A(16x32 f16) * B(32x16 f16) + C(16x16 f32)
  return __builtin_amdgcn_wmma_f32_16x16x32_f16(false, a, false, b, (short)0, c,
                                                false, false);
}

// xor-butterfly over the 16-lane half-wave group via ds_swizzle (wave32).
template <int XM>
__device__ __forceinline__ float swz_xor(float v) {
  int r = __builtin_amdgcn_ds_swizzle(__float_as_int(v), (XM << 10) | 0x1F);
  return __int_as_float(r);
}
__device__ __forceinline__ float grp16_max(float v) {
  v = fmaxf(v, swz_xor<1>(v));
  v = fmaxf(v, swz_xor<2>(v));
  v = fmaxf(v, swz_xor<4>(v));
  v = fmaxf(v, swz_xor<8>(v));
  return v;
}
__device__ __forceinline__ float grp16_sum(float v) {
  v += swz_xor<1>(v);
  v += swz_xor<2>(v);
  v += swz_xor<4>(v);
  v += swz_xor<8>(v);
  return v;
}

// ---------------------------------------------------------------------------
// TDM: issue a 2D tile load (64 rows x 64 halves) from Bt[N][K] (f16) into
// LDS at byte offset lds_off.  D# layout per CDNA5 ISA ch.8.
// ---------------------------------------------------------------------------
#if HAVE_TDM
__device__ __forceinline__ void tdm_load_tile_b(unsigned long long gaddr,
                                                unsigned int lds_off,
                                                unsigned int Kdim,
                                                unsigned int Ntot) {
  // group0: [1:0]=count=1, [63:32]=lds_addr, [120:64]=global_addr, [127:126]=2
  v4u g0;
  g0.x = 1u;                                   // count = 1 valid descriptor
  g0.y = lds_off;                              // LDS byte address
  g0.z = (unsigned int)gaddr;                  // global_addr[31:0]
  g0.w = ((unsigned int)(gaddr >> 32) & 0x1FFFFFFu) | (2u << 30);  // type=2
  // group1 (256b): data_size=1(2B); tensor_dim0=K, tensor_dim1=N;
  // tile_dim0=64, tile_dim1=64; tensor_dim0_stride=K.
  v8i g1;
  g1[0] = (int)(1u << 16);                                   // data_size=2B
  g1[1] = (int)((Kdim & 0xFFFFu) << 16);                     // tdim0[15:0]
  g1[2] = (int)((Kdim >> 16) | ((Ntot & 0xFFFFu) << 16));    // tdim0hi|tdim1lo
  g1[3] = (int)((Ntot >> 16) | (64u << 16));                 // tdim1hi|tile0=64
  g1[4] = (int)(64u);                                        // tile1=64,tile2=0
  g1[5] = (int)Kdim;                                         // dim0_stride lo32
  g1[6] = 0;
  g1[7] = 0;
  v4i z4 = {0, 0, 0, 0};
#if __clang_major__ >= 23
  v8i z8 = {0, 0, 0, 0, 0, 0, 0, 0};
  __builtin_amdgcn_tensor_load_to_lds(g0, g1, z4, z4, z8, 0);
#else
  __builtin_amdgcn_tensor_load_to_lds(g0, g1, z4, z4, 0);
#endif
}
#endif

// ---------------------------------------------------------------------------
// Elementwise helpers
// ---------------------------------------------------------------------------
__global__ void cvt_f32_to_f16(const float* __restrict__ in,
                               _Float16* __restrict__ out, int n) {
  int i = blockIdx.x * blockDim.x + threadIdx.x;
  if (i < n) out[i] = (_Float16)in[i];
}

// Wt[n][k] = (f16) W[k][n]   (W is [K][N] fp32)
__global__ void wtrans_f16(const float* __restrict__ W,
                           _Float16* __restrict__ Wt, int K, int N) {
  int i = blockIdx.x * blockDim.x + threadIdx.x;
  if (i >= K * N) return;
  int k = i % K, n = i / K;
  Wt[(size_t)n * K + k] = (_Float16)W[(size_t)k * N + n];
}

// ---------------------------------------------------------------------------
// RoPE (+ optional per-head q scaling).
// in : f16 [B][T][nH*64] ; out : f16 [B][nH][T][64]
// ---------------------------------------------------------------------------
__global__ void rope_kernel(const _Float16* __restrict__ in,
                            const float* __restrict__ hw,
                            _Float16* __restrict__ out, int nH) {
  int idx = blockIdx.x * blockDim.x + threadIdx.x;
  int total = Bn * Tn * nH * HDc;
  if (idx >= total) return;
  int d  = idx & 63;
  int h  = (idx >> 6) % nH;
  int t  = (idx >> 6) / nH % Tn;
  int b  = (idx >> 6) / nH / Tn;

  size_t base = ((size_t)(b * Tn + t)) * (nH * HDc) + (size_t)h * HDc;
  float xv = (float)in[base + d];
  int   dp = (d < 32) ? d + 32 : d - 32;
  float xr = (float)in[base + dp];
  if (d < 32) xr = -xr;

  int band = (d & 31) >> 1;                      // 0..15 freq band
  // inv_freq = 10000^(-band/16) = exp(-band * ln(10000)/16)
  float inv_freq = __expf(-0.57564627324851f * (float)band);
  float ang = (float)t * inv_freq;
  float val = xv * __cosf(ang) + xr * __sinf(ang);
  if (hw) val *= hw[h];

  out[((size_t)(b * nH + h) * Tn + t) * HDc + d] = (_Float16)val;
}

// Vt[b][kv][d][t] = V0[b][t][kv*64+d]
__global__ void vtrans_kernel(const _Float16* __restrict__ V0,
                              _Float16* __restrict__ Vt) {
  int idx = blockIdx.x * blockDim.x + threadIdx.x;
  int total = Bn * HKVn * HDc * Tn;
  if (idx >= total) return;
  int t  = idx % Tn;
  int d  = (idx / Tn) & 63;
  int kv = (idx / Tn / HDc) % HKVn;
  int b  = idx / (Tn * HDc * HKVn);
  Vt[idx] = V0[((size_t)(b * Tn + t)) * (HKVn * HDc) + kv * HDc + d];
}

// ---------------------------------------------------------------------------
// WMMA GEMM with TDM-staged B tiles:
//   C[M][N] = A[M][K](f16) @ Bt[N][K](f16)^T + bias
// grid = (M/64, N/64), block = 128 (4 waves). Wave w computes rows
// bx*64 + w*16 .. +15 across a 64-wide N tile (4 accumulator chains).
// The shared 64x64-half B tile is DMA'd into LDS (double buffered) by the
// Tensor Data Mover; A rows stream from global per wave.
// All 5 fragments are hoisted ahead of the 4 WMMAs so a single combined
// loadcnt/dscnt wait covers one K-step and the WMMAs issue back-to-back.
// ---------------------------------------------------------------------------
__device__ __forceinline__ void gemm_core(const _Float16* __restrict__ A,
                                          const _Float16* __restrict__ Bt,
                                          int N, int K, v8f acc[4],
                                          _Float16* lds) {
  const int tid  = threadIdx.x;
  const int lane = tid & 31, wv = tid >> 5;
  const int m16  = lane & 15, g = (lane >> 4) & 1;
  const int row0 = blockIdx.y * 64;              // B-tile first row (N dim)
  const int NC   = K / 64;                       // 64-half K chunks

  const _Float16* arow = A + (size_t)(blockIdx.x * 64 + wv * 16 + m16) * K;

#if HAVE_TDM
  const unsigned int lds_base = (unsigned int)(size_t)lds;
  const unsigned long long bt_base =
      (unsigned long long)(size_t)Bt + (unsigned long long)row0 * K * 2ull;
  // prologue: stage chunk 0 into buffer 0
  if (wv == 0)
    tdm_load_tile_b(bt_base, lds_base, (unsigned int)K, (unsigned int)N);
#endif

  for (int c = 0; c < NC; ++c) {
#if HAVE_TDM
    if (c + 1 < NC) {
      if (wv == 0) {
        tdm_load_tile_b(bt_base + (unsigned long long)(c + 1) * 128ull,
                        lds_base + (unsigned int)((c + 1) & 1) * 8192u,
                        (unsigned int)K, (unsigned int)N);
        __builtin_amdgcn_s_wait_tensorcnt(1);   // chunk c complete (in-order)
      }
    } else {
      if (wv == 0) __builtin_amdgcn_s_wait_tensorcnt(0);
    }
    __syncthreads();                            // hand staged tile to all waves
    const _Float16* bb = lds + (size_t)(c & 1) * 4096;
#else
    __syncthreads();                            // previous compute done
    for (int idx = tid; idx < 512; idx += 128) {
      int r = idx >> 3, off = (idx & 7) * 8;
      *(v8h*)(lds + r * 64 + off) =
          *(const v8h*)(Bt + (size_t)(row0 + r) * K + (size_t)c * 64 + off);
    }
    __syncthreads();
    const _Float16* bb = lds;
#endif
    if (c + 1 < NC) __builtin_prefetch(arow + (size_t)(c + 1) * 64, 0, 1);

#pragma unroll
    for (int ks = 0; ks < 64; ks += 32) {
      // hoist ALL fragment loads ahead of the WMMAs: one load clause,
      // one combined wait, then 4 independent back-to-back WMMAs.
      const _Float16* p0 = bb + ( 0 + m16) * 64 + ks + g * 16;
      const _Float16* p1 = bb + (16 + m16) * 64 + ks + g * 16;
      const _Float16* p2 = bb + (32 + m16) * 64 + ks + g * 16;
      const _Float16* p3 = bb + (48 + m16) * 64 + ks + g * 16;
      v16h f0 = cat8(*(const v8h*)p0, *(const v8h*)(p0 + 8));
      v16h f1 = cat8(*(const v8h*)p1, *(const v8h*)(p1 + 8));
      v16h f2 = cat8(*(const v8h*)p2, *(const v8h*)(p2 + 8));
      v16h f3 = cat8(*(const v8h*)p3, *(const v8h*)(p3 + 8));
      v16h a  = cat8(*(const v8h*)(arow + c * 64 + ks + g * 16),
                     *(const v8h*)(arow + c * 64 + ks + g * 16 + 8));
      acc[0] = wmma_f16(a, f0, acc[0]);
      acc[1] = wmma_f16(a, f1, acc[1]);
      acc[2] = wmma_f16(a, f2, acc[2]);
      acc[3] = wmma_f16(a, f3, acc[3]);
    }
    __syncthreads();                            // tile consumed; safe to reuse
  }
}

__global__ __launch_bounds__(128)
void gemm_f16out(const _Float16* __restrict__ A, const _Float16* __restrict__ Bt,
                 const float* __restrict__ bias, _Float16* __restrict__ C,
                 int N, int K) {
  __shared__ _Float16 Blds[2][64 * 64];
  v8f acc[4] = {};
  gemm_core(A, Bt, N, K, acc, &Blds[0][0]);
  int tid = threadIdx.x, lane = tid & 31, wv = tid >> 5;
  int m16 = lane & 15, g = (lane >> 4) & 1;
#pragma unroll
  for (int r = 0; r < 8; ++r) {
    int orow = blockIdx.x * 64 + wv * 16 + r + 8 * g;
#pragma unroll
    for (int t = 0; t < 4; ++t) {
      int ocol = blockIdx.y * 64 + t * 16 + m16;
      C[(size_t)orow * N + ocol] = (_Float16)(acc[t][r] + bias[ocol]);
    }
  }
}

__global__ __launch_bounds__(128)
void gemm_f32out(const _Float16* __restrict__ A, const _Float16* __restrict__ Bt,
                 const float* __restrict__ bias, float* __restrict__ C,
                 int N, int K) {
  __shared__ _Float16 Blds[2][64 * 64];
  v8f acc[4] = {};
  gemm_core(A, Bt, N, K, acc, &Blds[0][0]);
  int tid = threadIdx.x, lane = tid & 31, wv = tid >> 5;
  int m16 = lane & 15, g = (lane >> 4) & 1;
#pragma unroll
  for (int r = 0; r < 8; ++r) {
    int orow = blockIdx.x * 64 + wv * 16 + r + 8 * g;
#pragma unroll
    for (int t = 0; t < 4; ++t) {
      int ocol = blockIdx.y * 64 + t * 16 + m16;
      C[(size_t)orow * N + ocol] = acc[t][r] + bias[ocol];
    }
  }
}

// ---------------------------------------------------------------------------
// Flash attention with structural skip-mask sparsity.
// grid = (T/16, H, B), block = 32 (one wave = one 16-query tile).
// Qr [B][H][T][64], Kr [B][HKV][T][64], Vt [B][HKV][64][T],
// O  [B][T][H*64] (f16).
// ---------------------------------------------------------------------------
__global__ __launch_bounds__(32)
void flash_kernel(const _Float16* __restrict__ Qr,
                  const _Float16* __restrict__ Kr,
                  const _Float16* __restrict__ Vt,
                  _Float16* __restrict__ O) {
  const int qt = blockIdx.x, h = blockIdx.y, b = blockIdx.z;
  const int kv = h / NREP;
  const int lane = threadIdx.x, m16 = lane & 15, g = lane >> 4;

  __shared__ _Float16 pl[16 * 32];   // P tile staging (C-layout -> A-layout)

  // Q A-fragments (hd 0..31 and 32..63), loaded once.
  const _Float16* qb = Qr + (((size_t)(b * Hn + h)) * Tn + qt * 16 + m16) * HDc;
  v16h aq0 = cat8(*(const v8h*)(qb + g * 16),      *(const v8h*)(qb + g * 16 + 8));
  v16h aq1 = cat8(*(const v8h*)(qb + 32 + g * 16), *(const v8h*)(qb + 40 + g * 16));

  const _Float16* kb = Kr + ((size_t)(b * HKVn + kv)) * Tn * HDc;
  const _Float16* vb = Vt + ((size_t)(b * HKVn + kv)) * HDc * Tn;

  float mst[8], lst[8];
  v8f o[4] = {};
#pragma unroll
  for (int r = 0; r < 8; ++r) { mst[r] = -3.0e38f; lst[r] = 0.0f; }

  auto process = [&](int kblk) {
    const int k0 = kblk * 32;
    // ---- scores S[16 x 32]: hoist all K fragments, then 4 WMMAs
    const _Float16* kr0 = kb + (size_t)(k0 + m16) * HDc;
    const _Float16* kr1 = kb + (size_t)(k0 + 16 + m16) * HDc;
    v16h bk00 = cat8(*(const v8h*)(kr0 + g * 16),      *(const v8h*)(kr0 + g * 16 + 8));
    v16h bk01 = cat8(*(const v8h*)(kr0 + 32 + g * 16), *(const v8h*)(kr0 + 40 + g * 16));
    v16h bk10 = cat8(*(const v8h*)(kr1 + g * 16),      *(const v8h*)(kr1 + g * 16 + 8));
    v16h bk11 = cat8(*(const v8h*)(kr1 + 32 + g * 16), *(const v8h*)(kr1 + 40 + g * 16));
    v8f s[2];
    s[0] = wmma_f16(aq0, bk00, (v8f){});
    s[1] = wmma_f16(aq0, bk10, (v8f){});
    s[0] = wmma_f16(aq1, bk01, s[0]);
    s[1] = wmma_f16(aq1, bk11, s[1]);
    // ---- scale + additive skip mask (fp32)
    float sv[2][8], rowmax[8];
#pragma unroll
    for (int r = 0; r < 8; ++r) {
      int i = qt * 16 + r + 8 * g;
#pragma unroll
      for (int nb = 0; nb < 2; ++nb) {
        int j = k0 + nb * 16 + m16;
        bool ok = (abs(i - j) <= HALF_WIN) || (i < GLOB) || (j < GLOB);
        sv[nb][r] = s[nb][r] * SCALE + (ok ? 0.0f : NEG_INF);
      }
      rowmax[r] = grp16_max(fmaxf(sv[0][r], sv[1][r]));
    }
    // ---- online softmax update
    float alpha[8];
#pragma unroll
    for (int r = 0; r < 8; ++r) {
      float mn = fmaxf(mst[r], rowmax[r]);
      alpha[r] = __expf(mst[r] - mn);
      mst[r]   = mn;
      float ps = 0.0f;
#pragma unroll
      for (int nb = 0; nb < 2; ++nb) {
        float p = __expf(sv[nb][r] - mn);
        sv[nb][r] = p;
        ps += p;
      }
      lst[r] = lst[r] * alpha[r] + grp16_sum(ps);
    }
#pragma unroll
    for (int cb = 0; cb < 4; ++cb)
#pragma unroll
      for (int r = 0; r < 8; ++r) o[cb][r] *= alpha[r];

    // ---- P (C-layout) -> LDS -> A-layout fragment
#pragma unroll
    for (int r = 0; r < 8; ++r) {
      int row = r + 8 * g;
      pl[row * 32 +      m16] = (_Float16)sv[0][r];
      pl[row * 32 + 16 + m16] = (_Float16)sv[1][r];
    }
    __syncthreads();
    // hoist P fragment + all 4 V fragments, then 4 back-to-back WMMAs
    const _Float16* pr = pl + m16 * 32;
    v16h ap = cat8(*(const v8h*)(pr + g * 16), *(const v8h*)(pr + g * 16 + 8));
    const _Float16* vr0 = vb + (size_t)( 0 + m16) * Tn + k0 + g * 16;
    const _Float16* vr1 = vb + (size_t)(16 + m16) * Tn + k0 + g * 16;
    const _Float16* vr2 = vb + (size_t)(32 + m16) * Tn + k0 + g * 16;
    const _Float16* vr3 = vb + (size_t)(48 + m16) * Tn + k0 + g * 16;
    v16h bv0 = cat8(*(const v8h*)vr0, *(const v8h*)(vr0 + 8));
    v16h bv1 = cat8(*(const v8h*)vr1, *(const v8h*)(vr1 + 8));
    v16h bv2 = cat8(*(const v8h*)vr2, *(const v8h*)(vr2 + 8));
    v16h bv3 = cat8(*(const v8h*)vr3, *(const v8h*)(vr3 + 8));
    o[0] = wmma_f16(ap, bv0, o[0]);
    o[1] = wmma_f16(ap, bv1, o[1]);
    o[2] = wmma_f16(ap, bv2, o[2]);
    o[3] = wmma_f16(ap, bv3, o[3]);
    __syncthreads();
  };

  if (qt == 0) {
    for (int kblk = 0; kblk < Tn / 32; ++kblk) process(kblk);
  } else {
    process(0);  // global keys (j < 8) live in key-block 0
    int klo = (qt * 16 - HALF_WIN) / 32;
    int khi = (qt * 16 + 15 + HALF_WIN) / 32;
    if (khi > Tn / 32 - 1) khi = Tn / 32 - 1;
    for (int kblk = klo; kblk <= khi; ++kblk)
      if (kblk != 0) process(kblk);
  }

  // ---- normalize + store O as [B][T][H*64]
#pragma unroll
  for (int r = 0; r < 8; ++r) {
    float inv = 1.0f / lst[r];
    int i = qt * 16 + r + 8 * g;
    size_t base = ((size_t)(b * Tn + i)) * (Hn * HDc) + (size_t)h * HDc;
#pragma unroll
    for (int cb = 0; cb < 4; ++cb)
      O[base + cb * 16 + m16] = (_Float16)(o[cb][r] * inv);
  }
}

// ---------------------------------------------------------------------------
// Host-side orchestration
// ---------------------------------------------------------------------------
extern "C" void kernel_launch(void* const* d_in, const int* in_sizes, int n_in,
                              void* d_out, int out_size, void* d_ws, size_t ws_size,
                              hipStream_t stream) {
  (void)in_sizes; (void)n_in; (void)out_size; (void)ws_size;
  const float* x  = (const float*)d_in[0];
  const float* Wq = (const float*)d_in[1];
  const float* bq = (const float*)d_in[2];
  const float* Wk = (const float*)d_in[3];
  const float* bk = (const float*)d_in[4];
  const float* Wv = (const float*)d_in[5];
  const float* bv = (const float*)d_in[6];
  const float* Wo = (const float*)d_in[7];
  const float* bo = (const float*)d_in[8];
  const float* hw = (const float*)d_in[9];
  float* out = (float*)d_out;

  // workspace carve (256B aligned)
  size_t off = 0;
  auto carve = [&](size_t bytes) {
    void* p = (char*)d_ws + off;
    off += (bytes + 255) & ~(size_t)255;
    return p;
  };
  const int MT = Bn * Tn;  // 4096 rows
  _Float16* xh  = (_Float16*)carve((size_t)MT * Dn * 2);          // also reused as O
  _Float16* Wqt = (_Float16*)carve((size_t)Dn * Dn * 2);
  _Float16* Wkt = (_Float16*)carve((size_t)(HKVn * HDc) * Dn * 2);
  _Float16* Wvt = (_Float16*)carve((size_t)(HKVn * HDc) * Dn * 2);
  _Float16* Wot = (_Float16*)carve((size_t)Dn * Dn * 2);
  _Float16* Q0  = (_Float16*)carve((size_t)MT * Hn * HDc * 2);
  _Float16* K0  = (_Float16*)carve((size_t)MT * HKVn * HDc * 2);
  _Float16* V0  = (_Float16*)carve((size_t)MT * HKVn * HDc * 2);
  _Float16* Qr  = (_Float16*)carve((size_t)MT * Hn * HDc * 2);
  _Float16* Kr  = (_Float16*)carve((size_t)MT * HKVn * HDc * 2);
  _Float16* Vt  = (_Float16*)carve((size_t)MT * HKVn * HDc * 2);
  _Float16* Oa  = xh;  // x no longer needed after projections

  const int TPB = 256;
  auto blocks = [](int n, int tpb) { return (n + tpb - 1) / tpb; };

  // 1) fp32 -> f16 activations, transposed f16 weights
  cvt_f32_to_f16<<<blocks(MT * Dn, TPB), TPB, 0, stream>>>(x, xh, MT * Dn);
  wtrans_f16<<<blocks(Dn * Dn, TPB), TPB, 0, stream>>>(Wq, Wqt, Dn, Dn);
  wtrans_f16<<<blocks(Dn * HKVn * HDc, TPB), TPB, 0, stream>>>(Wk, Wkt, Dn, HKVn * HDc);
  wtrans_f16<<<blocks(Dn * HKVn * HDc, TPB), TPB, 0, stream>>>(Wv, Wvt, Dn, HKVn * HDc);
  wtrans_f16<<<blocks(Dn * Dn, TPB), TPB, 0, stream>>>(Wo, Wot, Dn, Dn);

  // 2) QKV projections (WMMA + TDM-staged weight tiles)
  gemm_f16out<<<dim3(MT / 64, Dn / 64), 128, 0, stream>>>(xh, Wqt, bq, Q0, Dn, Dn);
  gemm_f16out<<<dim3(MT / 64, (HKVn * HDc) / 64), 128, 0, stream>>>(xh, Wkt, bk, K0, HKVn * HDc, Dn);
  gemm_f16out<<<dim3(MT / 64, (HKVn * HDc) / 64), 128, 0, stream>>>(xh, Wvt, bv, V0, HKVn * HDc, Dn);

  // 3) RoPE + layout changes
  rope_kernel<<<blocks(Bn * Tn * Hn * HDc, TPB), TPB, 0, stream>>>(Q0, hw, Qr, Hn);
  rope_kernel<<<blocks(Bn * Tn * HKVn * HDc, TPB), TPB, 0, stream>>>(K0, nullptr, Kr, HKVn);
  vtrans_kernel<<<blocks(Bn * HKVn * HDc * Tn, TPB), TPB, 0, stream>>>(V0, Vt);

  // 4) masked flash attention (WMMA + LDS + ds_swizzle reductions)
  flash_kernel<<<dim3(Tn / 16, Hn, Bn), 32, 0, stream>>>(Qr, Kr, Vt, Oa);

  // 5) output projection (WMMA, fp32 out)
  gemm_f32out<<<dim3(MT / 64, Dn / 64), 128, 0, stream>>>(Oa, Wot, bo, out, Dn, Dn);
}
